// NetGINE_23682449670434
// MI455X (gfx1250) — compile-verified, matrix operand order checked by reading, not run on hardware
//
#include <hip/hip_runtime.h>
#include <hip/hip_bf16.h>

typedef __bf16        v16bf __attribute__((ext_vector_type(16)));
typedef float         v8f   __attribute__((ext_vector_type(8)));
typedef unsigned int  v8u   __attribute__((ext_vector_type(8)));

union Frag {
    v16bf v;
    unsigned short u[16];
    v8u i;
};

__device__ __forceinline__ unsigned short f2bf(float f) {
    unsigned int u = __float_as_uint(f);
    u += 0x7fffu + ((u >> 16) & 1u);   // round-to-nearest-even to bf16
    return (unsigned short)(u >> 16);
}

__device__ __forceinline__ v8f zero8() {
    v8f z = {0.f, 0.f, 0.f, 0.f, 0.f, 0.f, 0.f, 0.f};
    return z;
}

// ---------------------------------------------------------------------------
// prep: bf16 fragment-layout weights + folded BN scale/shift
// Fragment layout (matches CDNA5 16x16x32 bf16 A/B VGPR striping):
//   elem index e -> i16 = e&15, lane = (e>>4)&31, ft = e>>9, nt = ft%NT, kt = ft/NT
//   n = nt*16 + (lane&15)
//   k = kt*32 + (lane<16 ? 0 : 8) + i16 + (i16>=8 ? 8 : 0)
// ---------------------------------------------------------------------------
__global__ void prep_kernel(
    const float* w0, const float* w1, const float* w2,
    const float* w3, const float* w4, const float* w5,
    const float* w6, const float* w7, const float* w8,
    const float* g1, const float* b1, const float* m1, const float* v1,
    const float* g2, const float* b2, const float* m2, const float* v2,
    const float* g3, const float* b3, const float* m3, const float* v3,
    const float* g4, const float* b4, const float* m4, const float* v4,
    unsigned short* wf, float* bnbuf)
{
    const float* W[9] = {w0, w1, w2, w3, w4, w5, w6, w7, w8};
    const int KK[9]  = {28, 28, 28, 64, 64, 64, 64, 64, 64};
    const int NN[9]  = {28, 28, 64, 64, 64, 64, 64, 64, 64};
    const int KT[9]  = {1, 1, 1, 2, 2, 2, 2, 2, 2};
    const int NT[9]  = {2, 2, 4, 4, 4, 4, 4, 4, 4};
    const int OFF[9] = {0, 1024, 2048, 4096, 8192, 12288, 16384, 20480, 24576};

    int tid = blockIdx.x * blockDim.x + threadIdx.x;
    int stride = gridDim.x * blockDim.x;

    for (int wi = 0; wi < 9; ++wi) {
        int total = KT[wi] * NT[wi] * 512;
        for (int e = tid; e < total; e += stride) {
            int i16 = e & 15;
            int lane = (e >> 4) & 31;
            int ft = e >> 9;
            int nt = ft % NT[wi];
            int kt = ft / NT[wi];
            int n = nt * 16 + (lane & 15);
            int k = kt * 32 + ((lane < 16) ? 0 : 8) + i16 + ((i16 >= 8) ? 8 : 0);
            float val = (k < KK[wi] && n < NN[wi]) ? W[wi][k * NN[wi] + n] : 0.f;
            wf[OFF[wi] + e] = f2bf(val);
        }
    }

    const float* G[4] = {g1, g2, g3, g4};
    const float* B[4] = {b1, b2, b3, b4};
    const float* M[4] = {m1, m2, m3, m4};
    const float* V[4] = {v1, v2, v3, v4};
    for (int e = tid; e < 256; e += stride) {
        int l = e >> 6, c = e & 63;
        float sc = G[l][c] * rsqrtf(V[l][c] + 1e-5f);
        bnbuf[l * 128 + c] = sc;
        bnbuf[l * 128 + 64 + c] = B[l][c] - M[l][c] * sc;
    }
}

// ---------------------------------------------------------------------------
// edge kernel: e = relu(ea@be1)@be2 (WMMA), msg = relu(h[src]+e), agg[dst]+=msg
// One wave owns one 16-edge tile; 8 waves per block.
// ---------------------------------------------------------------------------
template<int KT, int NT>
__global__ __launch_bounds__(256)
void edge_kernel(const float* __restrict__ ea,
                 const int* __restrict__ src, const int* __restrict__ dst,
                 const float* __restrict__ be1,
                 const unsigned short* __restrict__ be2f,
                 const float* __restrict__ h, float* __restrict__ agg,
                 int E, int Dh, int De, int hstride)
{
    __shared__ unsigned short hidS[8][16 * 64];
    __shared__ float eS[8][16 * 64];
    __shared__ float eaS[8][48];

    int wave = threadIdx.x >> 5, lane = threadIdx.x & 31;
    long tile = (long)blockIdx.x * 8 + wave;
    long e0 = tile * 16;
    unsigned short* hid = hidS[wave];
    float* et = eS[wave];
    float* eat = eaS[wave];

    long rem = (long)E - e0;
    int rows = rem <= 0 ? 0 : (rem >= 16 ? 16 : (int)rem);

    // stage edge_attr tile [16 x 3]
    for (int j = lane; j < 48; j += 32) {
        int r = j / 3, k = j - r * 3;
        eat[j] = (r < rows) ? ea[(e0 + r) * 3 + k] : 0.f;
    }
    __syncthreads();

    // hidden = relu(ea @ be1), K=3 scalar stage -> bf16 LDS tile [16 x KT*32]
    for (int c = lane; c < KT * 32; c += 32) {
        if (c < Dh) {
            float q0 = be1[c], q1 = be1[Dh + c], q2 = be1[2 * Dh + c];
            for (int r = 0; r < 16; ++r) {
                float hv = fmaf(eat[r * 3], q0,
                           fmaf(eat[r * 3 + 1], q1, eat[r * 3 + 2] * q2));
                hid[r * 64 + c] = f2bf(fmaxf(hv, 0.f));
            }
        } else {
            for (int r = 0; r < 16; ++r) hid[r * 64 + c] = 0;
        }
    }
    __syncthreads();

    // e = hidden @ be2 via WMMA
    int arow = lane & 15;
    int kb0 = (lane < 16) ? 0 : 8;
    v8f acc[NT];
#pragma unroll
    for (int nt = 0; nt < NT; ++nt) acc[nt] = zero8();
#pragma unroll
    for (int kt = 0; kt < KT; ++kt) {
        Frag a;
        int kbase = kt * 32 + kb0;
#pragma unroll
        for (int i = 0; i < 16; ++i) {
            int k = kbase + i + ((i >= 8) ? 8 : 0);
            a.u[i] = hid[arow * 64 + k];
        }
#pragma unroll
        for (int nt = 0; nt < NT; ++nt) {
            Frag b;
            b.i = ((const v8u*)be2f)[(kt * NT + nt) * 32 + lane];
            acc[nt] = __builtin_amdgcn_wmma_f32_16x16x32_bf16(
                false, a.v, false, b.v, (short)0, acc[nt], false, false);
        }
    }
    int crow = (lane >> 4) * 8, ccol = lane & 15;
#pragma unroll
    for (int nt = 0; nt < NT; ++nt)
#pragma unroll
        for (int g = 0; g < 8; ++g)
            et[(crow + g) * 64 + nt * 16 + ccol] = acc[nt][g];
    __syncthreads();

    // fused gather + relu + scatter-add
    for (int r = 0; r < rows; ++r) {
        int s = src[e0 + r], d = dst[e0 + r];
        const float* hrow = h + (long)s * hstride;
        float* ar = agg + (long)d * 64;
        for (int c = lane; c < De; c += 32) {
            float m = fmaxf(hrow[c] + et[r * 64 + c], 0.f);
            atomicAdd(&ar[c], m);
        }
    }
}

// ---------------------------------------------------------------------------
// node kernel: z=(1+eps)h+agg; out = bn(relu(relu(z@m1)@m2))
// ---------------------------------------------------------------------------
template<int KT1, int NT1, int KT2>
__global__ __launch_bounds__(256)
void node_kernel(const float* __restrict__ hin, const float* __restrict__ agg,
                 const float* __restrict__ epsP,
                 const unsigned short* __restrict__ m1f,
                 const unsigned short* __restrict__ m2f,
                 const float* __restrict__ bnsc, const float* __restrict__ bnsh,
                 float* __restrict__ xout, int N, int Din, int hs)
{
    __shared__ unsigned short zS[8][16 * 64];
    __shared__ unsigned short tS[8][16 * 64];

    int wave = threadIdx.x >> 5, lane = threadIdx.x & 31;
    long tile = (long)blockIdx.x * 8 + wave;
    long n0 = tile * 16;
    unsigned short* z = zS[wave];
    unsigned short* t = tS[wave];

    long rem = (long)N - n0;
    int rows = rem <= 0 ? 0 : (rem >= 16 ? 16 : (int)rem);
    float eps1 = 1.f + epsP[0];

    for (int c = lane; c < 64; c += 32)
        for (int r = 0; r < 16; ++r) {
            float zv = 0.f;
            if (r < rows && c < Din)
                zv = eps1 * hin[(n0 + r) * hs + c] + agg[(n0 + r) * 64 + c];
            z[r * 64 + c] = f2bf(zv);
        }
    __syncthreads();

    int arow = lane & 15;
    int kb0 = (lane < 16) ? 0 : 8;
    int crow = (lane >> 4) * 8, ccol = lane & 15;

    // t = relu(z @ m1)
    {
        v8f acc[NT1];
#pragma unroll
        for (int nt = 0; nt < NT1; ++nt) acc[nt] = zero8();
#pragma unroll
        for (int kt = 0; kt < KT1; ++kt) {
            Frag a;
            int kbase = kt * 32 + kb0;
#pragma unroll
            for (int i = 0; i < 16; ++i) {
                int k = kbase + i + ((i >= 8) ? 8 : 0);
                a.u[i] = z[arow * 64 + k];
            }
#pragma unroll
            for (int nt = 0; nt < NT1; ++nt) {
                Frag b;
                b.i = ((const v8u*)m1f)[(kt * NT1 + nt) * 32 + lane];
                acc[nt] = __builtin_amdgcn_wmma_f32_16x16x32_bf16(
                    false, a.v, false, b.v, (short)0, acc[nt], false, false);
            }
        }
#pragma unroll
        for (int nt = 0; nt < NT1; ++nt)
#pragma unroll
            for (int g = 0; g < 8; ++g)
                t[(crow + g) * 64 + nt * 16 + ccol] = f2bf(fmaxf(acc[nt][g], 0.f));
    }
    __syncthreads();

    // out = t @ m2; relu; bn
    {
        v8f acc[4];
#pragma unroll
        for (int nt = 0; nt < 4; ++nt) acc[nt] = zero8();
#pragma unroll
        for (int kt = 0; kt < KT2; ++kt) {
            Frag a;
            int kbase = kt * 32 + kb0;
#pragma unroll
            for (int i = 0; i < 16; ++i) {
                int k = kbase + i + ((i >= 8) ? 8 : 0);
                a.u[i] = t[arow * 64 + k];
            }
#pragma unroll
            for (int nt = 0; nt < 4; ++nt) {
                Frag b;
                b.i = ((const v8u*)m2f)[(kt * 4 + nt) * 32 + lane];
                acc[nt] = __builtin_amdgcn_wmma_f32_16x16x32_bf16(
                    false, a.v, false, b.v, (short)0, acc[nt], false, false);
            }
        }
#pragma unroll
        for (int nt = 0; nt < 4; ++nt)
#pragma unroll
            for (int g = 0; g < 8; ++g) {
                int row = crow + g;
                if (row < rows) {
                    int col = nt * 16 + ccol;
                    float v = fmaxf(acc[nt][g], 0.f);
                    xout[(n0 + row) * 64 + col] = v * bnsc[col] + bnsh[col];
                }
            }
    }
}

// ---------------------------------------------------------------------------
// mean pool: atomic sum of cat([x1..x4]) into [G,256] + counts
// ---------------------------------------------------------------------------
__global__ __launch_bounds__(256)
void pool_kernel(const float* __restrict__ x1, const float* __restrict__ x2,
                 const float* __restrict__ x3, const float* __restrict__ x4,
                 const int* __restrict__ batch,
                 float* __restrict__ pooled, float* __restrict__ cnt, int N)
{
    long idx = (long)blockIdx.x * blockDim.x + threadIdx.x;
    long total = (long)N * 256;
    if (idx >= total) return;
    int n = (int)(idx >> 8);
    int c = (int)(idx & 255);
    int which = c >> 6, cc = c & 63;
    const float* p = (which == 0) ? x1 : (which == 1) ? x2 : (which == 2) ? x3 : x4;
    float v = p[(long)n * 64 + cc];
    int g = batch[n];
    atomicAdd(&pooled[(long)g * 256 + c], v);
    if (c == 0) atomicAdd(&cnt[g], 1.f);
}

// ---------------------------------------------------------------------------
// FC head: one 64-thread block per graph
// ---------------------------------------------------------------------------
__global__ __launch_bounds__(64)
void fc_kernel(const float* __restrict__ pooled, const float* __restrict__ cnt,
               const float* __restrict__ w1, const float* __restrict__ b1,
               const float* __restrict__ w2, const float* __restrict__ b2,
               const float* __restrict__ w3, const float* __restrict__ b3,
               const float* __restrict__ w4, const float* __restrict__ b4,
               float* __restrict__ out)
{
    __shared__ float A[256];
    __shared__ float Bf[64];
    int g = blockIdx.x, t = threadIdx.x;
    float inv = 1.f / fmaxf(cnt[g], 1.f);
    for (int i = t; i < 256; i += 64) A[i] = pooled[(long)g * 256 + i] * inv;
    __syncthreads();

    float a = b1[t];
    for (int k = 0; k < 256; ++k) a = fmaf(A[k], w1[k * 64 + t], a);
    Bf[t] = fmaxf(a, 0.f);
    __syncthreads();

    a = b2[t];
    for (int k = 0; k < 64; ++k) a = fmaf(Bf[k], w2[k * 64 + t], a);
    __syncthreads();
    A[t] = fmaxf(a, 0.f);
    __syncthreads();

    a = b3[t];
    for (int k = 0; k < 64; ++k) a = fmaf(A[k], w3[k * 64 + t], a);
    __syncthreads();
    Bf[t] = fmaxf(a, 0.f);
    __syncthreads();

    if (t == 0) {
        float s = b4[0];
        for (int k = 0; k < 64; ++k) s = fmaf(Bf[k], w4[k], s);
        out[g] = s;
    }
}

// ---------------------------------------------------------------------------
extern "C" void kernel_launch(void* const* d_in, const int* in_sizes, int n_in,
                              void* d_out, int out_size, void* d_ws, size_t ws_size,
                              hipStream_t stream)
{
    const float* x       = (const float*)d_in[0];
    const float* ea      = (const float*)d_in[1];
    const float* c1_be1  = (const float*)d_in[2];
    const float* c1_be2  = (const float*)d_in[3];
    const float* c1_m1   = (const float*)d_in[4];
    const float* c1_m2   = (const float*)d_in[5];
    const float* c1_eps  = (const float*)d_in[6];
    const float* c2_be1  = (const float*)d_in[7];
    const float* c2_be2  = (const float*)d_in[8];
    const float* c2_m1   = (const float*)d_in[9];
    const float* c2_m2   = (const float*)d_in[10];
    const float* c2_eps  = (const float*)d_in[11];
    const float* c3_be1  = (const float*)d_in[12];
    const float* c3_be2  = (const float*)d_in[13];
    const float* c3_m1   = (const float*)d_in[14];
    const float* c3_m2   = (const float*)d_in[15];
    const float* c3_eps  = (const float*)d_in[16];
    // bn: 17..32
    const float* fc1_w = (const float*)d_in[33];
    const float* fc1_b = (const float*)d_in[34];
    const float* fc2_w = (const float*)d_in[35];
    const float* fc2_b = (const float*)d_in[36];
    const float* fc3_w = (const float*)d_in[37];
    const float* fc3_b = (const float*)d_in[38];
    const float* fc4_w = (const float*)d_in[39];
    const float* fc4_b = (const float*)d_in[40];
    const int* eidx  = (const int*)d_in[41];
    const int* batch = (const int*)d_in[42];

    int N = in_sizes[0] / 28;
    int E = in_sizes[1] / 3;
    int G = out_size;
    const int* src = eidx;
    const int* dst = eidx + E;

    char* ws = (char*)d_ws;
    unsigned short* wf = (unsigned short*)ws;     // 57344 B of bf16 fragments
    float* bnbuf = (float*)(ws + 57344);          // 2048 B (scale/shift x4)
    size_t xbytes = (size_t)N * 64 * sizeof(float);
    float* agg = (float*)(ws + 65536);
    float* x1 = (float*)((char*)agg + xbytes);
    float* x2 = x1 + (size_t)N * 64;
    float* x3 = x2 + (size_t)N * 64;
    float* x4 = x3 + (size_t)N * 64;
    float* pooled = x4 + (size_t)N * 64;
    float* cnt = pooled + (size_t)G * 256;

    prep_kernel<<<32, 256, 0, stream>>>(
        c1_be2, c1_m1, c1_m2, c2_be2, c2_m1, c2_m2, c3_be2, c3_m1, c3_m2,
        (const float*)d_in[17], (const float*)d_in[18], (const float*)d_in[19], (const float*)d_in[20],
        (const float*)d_in[21], (const float*)d_in[22], (const float*)d_in[23], (const float*)d_in[24],
        (const float*)d_in[25], (const float*)d_in[26], (const float*)d_in[27], (const float*)d_in[28],
        (const float*)d_in[29], (const float*)d_in[30], (const float*)d_in[31], (const float*)d_in[32],
        wf, bnbuf);

    int tilesE = (E + 15) / 16;
    int gridE = (tilesE + 7) / 8;
    int tilesN = (N + 15) / 16;
    int gridN = (tilesN + 7) / 8;

    // ---- layer 1 (dims 28, padded to 32) ----
    hipMemsetAsync(agg, 0, xbytes, stream);
    edge_kernel<1, 2><<<gridE, 256, 0, stream>>>(
        ea, src, dst, c1_be1, wf + 0, x, agg, E, 28, 28, 28);
    node_kernel<1, 2, 1><<<gridN, 256, 0, stream>>>(
        x, agg, c1_eps, wf + 1024, wf + 2048, bnbuf + 0, bnbuf + 64, x1, N, 28, 28);

    // ---- layer 2 ----
    hipMemsetAsync(agg, 0, xbytes, stream);
    edge_kernel<2, 4><<<gridE, 256, 0, stream>>>(
        ea, src, dst, c2_be1, wf + 4096, x1, agg, E, 64, 64, 64);
    node_kernel<2, 4, 2><<<gridN, 256, 0, stream>>>(
        x1, agg, c2_eps, wf + 8192, wf + 12288, bnbuf + 128, bnbuf + 192, x2, N, 64, 64);

    // ---- layer 3 ----
    hipMemsetAsync(agg, 0, xbytes, stream);
    edge_kernel<2, 4><<<gridE, 256, 0, stream>>>(
        ea, src, dst, c3_be1, wf + 16384, x2, agg, E, 64, 64, 64);
    node_kernel<2, 4, 2><<<gridN, 256, 0, stream>>>(
        x2, agg, c3_eps, wf + 20480, wf + 24576, bnbuf + 256, bnbuf + 320, x3, N, 64, 64);

    // ---- layer 4 (conv3 weights reused, bn4) ----
    hipMemsetAsync(agg, 0, xbytes, stream);
    edge_kernel<2, 4><<<gridE, 256, 0, stream>>>(
        ea, src, dst, c3_be1, wf + 16384, x3, agg, E, 64, 64, 64);
    node_kernel<2, 4, 2><<<gridN, 256, 0, stream>>>(
        x3, agg, c3_eps, wf + 20480, wf + 24576, bnbuf + 384, bnbuf + 448, x4, N, 64, 64);

    // ---- pooling + FC head ----
    hipMemsetAsync(pooled, 0, ((size_t)G * 256 + (size_t)G) * sizeof(float), stream);
    long poolThreads = (long)N * 256;
    int gridP = (int)((poolThreads + 255) / 256);
    pool_kernel<<<gridP, 256, 0, stream>>>(x1, x2, x3, x4, batch, pooled, cnt, N);

    fc_kernel<<<G, 64, 0, stream>>>(pooled, cnt, fc1_w, fc1_b, fc2_w, fc2_b,
                                    fc3_w, fc3_b, fc4_w, fc4_b, (float*)d_out);
}
